// BalancedDevignModel_70789650972768
// MI455X (gfx1250) — compile-verified, matrix operand order checked by reading, not run on hardware
//
#include <hip/hip_runtime.h>
#include <math.h>

// ---------------------------------------------------------------------------
// CDNA5 (gfx1250) GatedGraphConv forward.
// Heavy GEMMs use v_wmma_f32_16x16x32_bf16 (bf16 operands, f32 accumulate).
// Edge scatter-add uses global f32 atomics (agg = 102MB, L2-resident).
// ---------------------------------------------------------------------------

typedef __attribute__((ext_vector_type(16))) __bf16 v16bf;
typedef __attribute__((ext_vector_type(8)))  float  v8f;

#define HDIM 256
#define NGRAPH 64

// K index inside a 32-wide K tile for fragment element e (0..15), given the
// lane half (lane>>4).  Matches the CDNA5 16-bit A/B operand layout:
//   VGPR v (=e>>1): K = (v<4 ? 0 : 16) + half*8 + (v&3)*2 + (e&1)
__device__ __forceinline__ int frag_k(int e, int half) {
    int v = e >> 1;
    return ((v < 4) ? 0 : 16) + half * 8 + (v & 3) * 2 + (e & 1);
}

__device__ __forceinline__ v8f wmma_bf16(v16bf a, v16bf b, v8f c) {
    return __builtin_amdgcn_wmma_f32_16x16x32_bf16(false, a, false, b,
                                                   (short)0, c, false, false);
}

__device__ __forceinline__ v16bf load_bfrag(const __bf16* __restrict__ pk,
                                            int nt_total_idx, int lane) {
    return *(const v16bf*)(pk + (((size_t)nt_total_idx * 32 + lane) << 4));
}

// ---------------------------------------------------------------------------
// Weight pre-pack: f32 matrix -> bf16 WMMA B fragments.
// pk layout: [kt][nt][lane(32)][e(16)]   (index == tid)
// transposed==0 : B[k][n] = W[k*ldW + n]
// transposed==1 : B[k][n] = W[n*ldW + k]   (for  X @ W^T)
// ---------------------------------------------------------------------------
__global__ void pack_b_kernel(const float* __restrict__ W, __bf16* __restrict__ pk,
                              int Krows, int Ncols, int ldW, int transposed,
                              int NT, int total) {
    int tid = blockIdx.x * blockDim.x + threadIdx.x;
    if (tid >= total) return;
    int e    = tid & 15;
    int lane = (tid >> 4) & 31;
    int rest = tid >> 9;
    int nt   = rest % NT;
    int kt   = rest / NT;
    int half = lane >> 4;
    int n = nt * 16 + (lane & 15);
    int k = kt * 32 + frag_k(e, half);
    float v = 0.0f;
    if (k < Krows && n < Ncols)
        v = transposed ? W[(size_t)n * ldW + k] : W[(size_t)k * ldW + n];
    pk[tid] = (__bf16)v;
}

// ---------------------------------------------------------------------------
// Input GEMM: Y[N,256] = X[N,D] @ Win[D,256] + b   (D<=128, zero padded)
// One wave per 16-row strip; two column tiles per iteration for WMMA ILP.
// ---------------------------------------------------------------------------
__global__ void __launch_bounds__(256)
gemm_in_kernel(const float* __restrict__ X, const __bf16* __restrict__ pk,
               const float* __restrict__ bias, float* __restrict__ Y,
               int N, int D, int nStrips) {
    int wave = (blockIdx.x * blockDim.x + threadIdx.x) >> 5;
    int lane = threadIdx.x & 31;
    if (wave >= nStrips) return;
    int half = lane >> 4, n16 = lane & 15;
    int arow = wave * 16 + n16; if (arow >= N) arow = N - 1;
    const float* xrow = X + (size_t)arow * D;

    v16bf af[4];
#pragma unroll
    for (int kt = 0; kt < 4; ++kt) {
        v16bf t;
#pragma unroll
        for (int e = 0; e < 16; ++e) {
            int k = kt * 32 + frag_k(e, half);
            t[e] = (__bf16)((k < D) ? xrow[k] : 0.0f);
        }
        af[kt] = t;
    }
    int row0 = wave * 16;
    bool full = (row0 + 16) <= N;                 // wave-uniform
    for (int jt = 0; jt < 16; jt += 2) {
        v8f acc0 = {}, acc1 = {};
#pragma unroll
        for (int kt = 0; kt < 4; ++kt) {
            v16bf b0 = load_bfrag(pk, kt * 16 + jt,     lane);
            v16bf b1 = load_bfrag(pk, kt * 16 + jt + 1, lane);
            acc0 = wmma_bf16(af[kt], b0, acc0);
            acc1 = wmma_bf16(af[kt], b1, acc1);
        }
        int col0 = jt * 16 + n16;
        float bb0 = bias[col0], bb1 = bias[col0 + 16];
        if (full) {
#pragma unroll
            for (int r = 0; r < 8; ++r) {
                size_t o = (size_t)(row0 + r + 8 * half) * HDIM + col0;
                Y[o]      = acc0[r] + bb0;
                Y[o + 16] = acc1[r] + bb1;
            }
        } else {
#pragma unroll
            for (int r = 0; r < 8; ++r) {
                int row = row0 + r + 8 * half;
                if (row < N) {
                    size_t o = (size_t)row * HDIM + col0;
                    Y[o]      = acc0[r] + bb0;
                    Y[o + 16] = acc1[r] + bb1;
                }
            }
        }
    }
}

// ---------------------------------------------------------------------------
// Square GEMM: Y[N,256] = A[N,256] @ B[256,256]  (B pre-packed bf16)
// ---------------------------------------------------------------------------
__global__ void __launch_bounds__(256)
gemm_h_kernel(const float* __restrict__ A, const __bf16* __restrict__ pk,
              float* __restrict__ Y, int N, int nStrips) {
    int wave = (blockIdx.x * blockDim.x + threadIdx.x) >> 5;
    int lane = threadIdx.x & 31;
    if (wave >= nStrips) return;
    int half = lane >> 4, n16 = lane & 15;
    int arow = wave * 16 + n16; if (arow >= N) arow = N - 1;
    const float* ar = A + (size_t)arow * HDIM;

    v16bf af[8];
#pragma unroll
    for (int kt = 0; kt < 8; ++kt) {
        v16bf t;
#pragma unroll
        for (int e = 0; e < 16; ++e) t[e] = (__bf16)ar[kt * 32 + frag_k(e, half)];
        af[kt] = t;
    }
    int row0 = wave * 16;
    bool full = (row0 + 16) <= N;                 // wave-uniform
    for (int jt = 0; jt < 16; jt += 2) {
        v8f acc0 = {}, acc1 = {};
#pragma unroll
        for (int kt = 0; kt < 8; ++kt) {
            v16bf b0 = load_bfrag(pk, kt * 16 + jt,     lane);
            v16bf b1 = load_bfrag(pk, kt * 16 + jt + 1, lane);
            acc0 = wmma_bf16(af[kt], b0, acc0);
            acc1 = wmma_bf16(af[kt], b1, acc1);
        }
        int col0 = jt * 16 + n16;
        if (full) {
#pragma unroll
            for (int r = 0; r < 8; ++r) {
                size_t o = (size_t)(row0 + r + 8 * half) * HDIM + col0;
                Y[o]      = acc0[r];
                Y[o + 16] = acc1[r];
            }
        } else {
#pragma unroll
            for (int r = 0; r < 8; ++r) {
                int row = row0 + r + 8 * half;
                if (row < N) {
                    size_t o = (size_t)row * HDIM + col0;
                    Y[o]      = acc0[r];
                    Y[o + 16] = acc1[r];
                }
            }
        }
    }
}

// ---------------------------------------------------------------------------
// Edge scatter-add: agg[dst[e]] += m[src[e]]  (one wave-slice of 32 lanes
// per edge, 8 channels per lane; agg is L2-resident -> fast f32 atomics)
// ---------------------------------------------------------------------------
__global__ void __launch_bounds__(256)
scatter_kernel(const float* __restrict__ m, const int* __restrict__ src,
               const int* __restrict__ dst, float* __restrict__ agg, int E) {
    size_t tid = (size_t)blockIdx.x * blockDim.x + threadIdx.x;
    size_t edge = tid >> 5;
    int lane = (int)(tid & 31);
    if (edge >= (size_t)E) return;
    int s = src[edge], d = dst[edge];
    const float* ms = m + (size_t)s * HDIM + lane * 8;
    float* ad = agg + (size_t)d * HDIM + lane * 8;
#pragma unroll
    for (int i = 0; i < 8; ++i) atomicAdd(&ad[i], ms[i]);
}

// ---------------------------------------------------------------------------
// Fused GRU step: gi = agg@Wih^T + bih, gh = h@Whh^T + bhh, gate math,
// in-place h update.  Per strip: 16 col tiles x 8 K tiles x 6 WMMA.
// The 6 accumulator chains are independent -> 6 back-to-back WMMAs per kt
// while the next kt's B-fragment clause is in flight.
// ---------------------------------------------------------------------------
__global__ void __launch_bounds__(256)
gru_kernel(float* __restrict__ h, const float* __restrict__ agg,
           const __bf16* __restrict__ pkWih, const __bf16* __restrict__ pkWhh,
           const float* __restrict__ bih, const float* __restrict__ bhh,
           int N, int nStrips) {
    int wave = (blockIdx.x * blockDim.x + threadIdx.x) >> 5;
    int lane = threadIdx.x & 31;
    if (wave >= nStrips) return;
    int half = lane >> 4, n16 = lane & 15;
    int arow = wave * 16 + n16; if (arow >= N) arow = N - 1;
    const float* hrow = h   + (size_t)arow * HDIM;
    const float* grow = agg + (size_t)arow * HDIM;

    v16bf ah[8], ag[8];
#pragma unroll
    for (int kt = 0; kt < 8; ++kt) {
        v16bf th, tg;
#pragma unroll
        for (int e = 0; e < 16; ++e) {
            int k = kt * 32 + frag_k(e, half);
            th[e] = (__bf16)hrow[k];
            tg[e] = (__bf16)grow[k];
        }
        ah[kt] = th; ag[kt] = tg;
    }
    int row0 = wave * 16;
    bool full = (row0 + 16) <= N;                 // wave-uniform
    for (int jt = 0; jt < 16; ++jt) {
        v8f ir = {}, iz = {}, inn = {}, hr = {}, hz = {}, hn = {};
#pragma unroll
        for (int kt = 0; kt < 8; ++kt) {
            int base = kt * 48;  // [kt][nt(48)][lane][16]
            v16bf bir = load_bfrag(pkWih, base + jt,      lane);
            v16bf biz = load_bfrag(pkWih, base + 16 + jt, lane);
            v16bf bin = load_bfrag(pkWih, base + 32 + jt, lane);
            v16bf bhr = load_bfrag(pkWhh, base + jt,      lane);
            v16bf bhz = load_bfrag(pkWhh, base + 16 + jt, lane);
            v16bf bhn = load_bfrag(pkWhh, base + 32 + jt, lane);
            ir  = wmma_bf16(ag[kt], bir, ir);
            iz  = wmma_bf16(ag[kt], biz, iz);
            inn = wmma_bf16(ag[kt], bin, inn);
            hr  = wmma_bf16(ah[kt], bhr, hr);
            hz  = wmma_bf16(ah[kt], bhz, hz);
            hn  = wmma_bf16(ah[kt], bhn, hn);
        }
        int col = jt * 16 + n16;
        float bi_r = bih[col], bi_z = bih[col + 256], bi_n = bih[col + 512];
        float bh_r = bhh[col], bh_z = bhh[col + 256], bh_n = bhh[col + 512];
#pragma unroll
        for (int r = 0; r < 8; ++r) {
            int row = row0 + r + 8 * half;
            if (!full && row >= N) continue;
            size_t o = (size_t)row * HDIM + col;
            float xr = ir[r] + bi_r + hr[r] + bh_r;
            float xz = iz[r] + bi_z + hz[r] + bh_z;
            float rg = 1.0f / (1.0f + __expf(-xr));
            float zg = 1.0f / (1.0f + __expf(-xz));
            float ng = tanhf(inn[r] + bi_n + rg * (hn[r] + bh_n));
            float hold = h[o];
            h[o] = (1.0f - zg) * ng + zg * hold;
        }
    }
}

// ---------------------------------------------------------------------------
// BatchNorm (training stats) over N rows x 256 cols, two-pass.
// ---------------------------------------------------------------------------
__global__ void colstats_kernel(const float* __restrict__ Y, int N,
                                float* __restrict__ sum, float* __restrict__ sumsq,
                                int rowsPerBlock) {
    int c  = threadIdx.x;                 // 256 cols
    int r0 = blockIdx.x * rowsPerBlock;
    int r1 = min(N, r0 + rowsPerBlock);
    float s = 0.0f, s2 = 0.0f;
    for (int r = r0; r < r1; ++r) {
        float v = Y[(size_t)r * HDIM + c];
        s += v; s2 += v * v;
    }
    atomicAdd(&sum[c], s);
    atomicAdd(&sumsq[c], s2);
}

__global__ void bnrelu_kernel(const float* __restrict__ Y, float* __restrict__ H,
                              const float* __restrict__ sum, const float* __restrict__ sumsq,
                              const float* __restrict__ g, const float* __restrict__ b,
                              int N) {
    size_t idx = (size_t)blockIdx.x * blockDim.x + threadIdx.x;
    if (idx >= (size_t)N * HDIM) return;
    int c = (int)(idx & (HDIM - 1));
    float invN = 1.0f / (float)N;
    float mean = sum[c] * invN;
    float var  = sumsq[c] * invN - mean * mean;
    float v = (Y[idx] - mean) * rsqrtf(var + 1e-5f) * g[c] + b[c];
    H[idx] = fmaxf(v, 0.0f);
}

// ---------------------------------------------------------------------------
// Per-graph pooling (mean + max).  h2 >= 0 (post-ReLU) so float-max can be
// done as unsigned-int atomicMax on the raw bits with init 0.
// ---------------------------------------------------------------------------
__global__ void count_kernel(const int* __restrict__ batch, float* __restrict__ counts, int N) {
    int n = blockIdx.x * blockDim.x + threadIdx.x;
    if (n < N) atomicAdd(&counts[batch[n]], 1.0f);
}

__global__ void pool_kernel(const float* __restrict__ H2, const int* __restrict__ batch,
                            float* __restrict__ gsum, unsigned int* __restrict__ gmax, int N) {
    size_t idx = (size_t)blockIdx.x * blockDim.x + threadIdx.x;
    if (idx >= (size_t)N * HDIM) return;
    int n = (int)(idx >> 8);
    int c = (int)(idx & (HDIM - 1));
    int g = batch[n];
    float v = H2[idx];
    atomicAdd(&gsum[g * HDIM + c], v);
    atomicMax(&gmax[g * HDIM + c], __float_as_uint(v));
}

__global__ void gcat_kernel(const float* __restrict__ gsum, const unsigned int* __restrict__ gmax,
                            const float* __restrict__ counts, float* __restrict__ gcat) {
    int idx = blockIdx.x * blockDim.x + threadIdx.x;
    if (idx >= NGRAPH * 2 * HDIM) return;
    int g = idx >> 9;
    int c = idx & 511;
    float cnt = counts[g];
    float v;
    if (c < HDIM) v = gsum[g * HDIM + c] / fmaxf(cnt, 1.0f);
    else          v = (cnt > 0.0f) ? __uint_as_float(gmax[g * HDIM + (c - HDIM)]) : 0.0f;
    gcat[idx] = v;
}

// ---------------------------------------------------------------------------
// Tiny MLP head on [64, .] — negligible work, plain kernels.
// ---------------------------------------------------------------------------
__global__ void fc_kernel(const float* __restrict__ in, const float* __restrict__ W,
                          const float* __restrict__ b, float* __restrict__ out,
                          int K, int J) {
    int idx = blockIdx.x * blockDim.x + threadIdx.x;
    if (idx >= NGRAPH * J) return;
    int g = idx / J, j = idx % J;
    float acc = b[j];
    for (int k = 0; k < K; ++k) acc += in[g * K + k] * W[k * J + j];
    out[idx] = acc;
}

__global__ void smallnorm_kernel(float* __restrict__ buf, const float* __restrict__ g,
                                 const float* __restrict__ b, int C) {
    int c = blockIdx.x * blockDim.x + threadIdx.x;
    if (c >= C) return;
    float s = 0.0f, s2 = 0.0f;
    for (int r = 0; r < NGRAPH; ++r) { float v = buf[r * C + c]; s += v; s2 += v * v; }
    float mean = s / (float)NGRAPH;
    float var  = s2 / (float)NGRAPH - mean * mean;
    float sc = g[c] * rsqrtf(var + 1e-5f);
    float sh = b[c] - mean * sc;
    for (int r = 0; r < NGRAPH; ++r)
        buf[r * C + c] = fmaxf(buf[r * C + c] * sc + sh, 0.0f);
}

// ---------------------------------------------------------------------------
extern "C" void kernel_launch(void* const* d_in, const int* in_sizes, int n_in,
                              void* d_out, int out_size, void* d_ws, size_t ws_size,
                              hipStream_t stream) {
    const float* x        = (const float*)d_in[0];
    const int*   ei       = (const int*)  d_in[1];
    const int*   batch    = (const int*)  d_in[2];
    const float* w_in     = (const float*)d_in[3];
    const float* b_in     = (const float*)d_in[4];
    const float* bn1_g    = (const float*)d_in[5];
    const float* bn1_b    = (const float*)d_in[6];
    const float* ggc_w    = (const float*)d_in[7];
    const float* gru_wih  = (const float*)d_in[8];
    const float* gru_whh  = (const float*)d_in[9];
    const float* gru_bih  = (const float*)d_in[10];
    const float* gru_bhh  = (const float*)d_in[11];
    const float* bn2_g    = (const float*)d_in[12];
    const float* bn2_b    = (const float*)d_in[13];
    const float* fc1_w    = (const float*)d_in[14];
    const float* fc1_b    = (const float*)d_in[15];
    const float* bn3_g    = (const float*)d_in[16];
    const float* bn3_b    = (const float*)d_in[17];
    const float* fc2_w    = (const float*)d_in[18];
    const float* fc2_b    = (const float*)d_in[19];
    const float* bn4_g    = (const float*)d_in[20];
    const float* bn4_b    = (const float*)d_in[21];
    const float* fc3_w    = (const float*)d_in[22];
    const float* fc3_b    = (const float*)d_in[23];

    const int N = in_sizes[2];
    const int E = in_sizes[1] / 2;
    const int D = in_sizes[0] / N;
    const int* src = ei;
    const int* dst = ei + E;

    // ---- workspace carve-up -------------------------------------------------
    char* w = (char*)d_ws;
    size_t off = 0;
    auto take = [&](size_t bytes) -> void* {
        void* p = w + off;
        off += (bytes + 255) & ~(size_t)255;
        return p;
    };
    float*  bufY   = (float*) take((size_t)N * HDIM * 4);
    float*  bufH   = (float*) take((size_t)N * HDIM * 4);
    float*  bufM   = (float*) take((size_t)N * HDIM * 4);
    float*  bufAgg = (float*) take((size_t)N * HDIM * 4);
    __bf16* pkWin  = (__bf16*)take((size_t)4 * 16 * 512 * 2);
    __bf16* pkGgc  = (__bf16*)take((size_t)5 * 8 * 16 * 512 * 2);
    __bf16* pkWih  = (__bf16*)take((size_t)8 * 48 * 512 * 2);
    __bf16* pkWhh  = (__bf16*)take((size_t)8 * 48 * 512 * 2);
    float*  sumB   = (float*) take(HDIM * 4);
    float*  sqB    = (float*) take(HDIM * 4);
    float*  counts = (float*) take(NGRAPH * 4);
    float*  gsum   = (float*) take((size_t)NGRAPH * HDIM * 4);
    unsigned int* gmax = (unsigned int*)take((size_t)NGRAPH * HDIM * 4);
    float*  gcat   = (float*) take((size_t)NGRAPH * 2 * HDIM * 4);
    float*  t1     = (float*) take((size_t)NGRAPH * HDIM * 4);
    float*  t2     = (float*) take((size_t)NGRAPH * 128 * 4);
    (void)ws_size; (void)n_in; (void)out_size;

    const int nStrips  = (N + 15) / 16;
    const int gemmBlks = (nStrips + 7) / 8;       // 8 waves / 256-thread block

    // ---- 1. pack weights into bf16 WMMA fragments ---------------------------
    {
        int tot = 4 * 16 * 512;
        pack_b_kernel<<<(tot + 255) / 256, 256, 0, stream>>>(w_in, pkWin, D, HDIM, HDIM, 0, 16, tot);
    }
    for (int i = 0; i < 5; ++i) {
        int tot = 8 * 16 * 512;
        pack_b_kernel<<<(tot + 255) / 256, 256, 0, stream>>>(
            ggc_w + (size_t)i * HDIM * HDIM, pkGgc + (size_t)i * 8 * 16 * 512,
            HDIM, HDIM, HDIM, 0, 16, tot);
    }
    {
        int tot = 8 * 48 * 512;
        pack_b_kernel<<<(tot + 255) / 256, 256, 0, stream>>>(gru_wih, pkWih, HDIM, 3 * HDIM, HDIM, 1, 48, tot);
        pack_b_kernel<<<(tot + 255) / 256, 256, 0, stream>>>(gru_whh, pkWhh, HDIM, 3 * HDIM, HDIM, 1, 48, tot);
    }

    // ---- 2. input layer: y = x @ w_in + b_in, BN1, ReLU ---------------------
    gemm_in_kernel<<<gemmBlks, 256, 0, stream>>>(x, pkWin, b_in, bufY, N, D, nStrips);
    hipMemsetAsync(sumB, 0, HDIM * 4, stream);
    hipMemsetAsync(sqB,  0, HDIM * 4, stream);
    colstats_kernel<<<(N + 511) / 512, 256, 0, stream>>>(bufY, N, sumB, sqB, 512);
    {
        size_t tot = (size_t)N * HDIM;
        bnrelu_kernel<<<(unsigned)((tot + 255) / 256), 256, 0, stream>>>(
            bufY, bufH, sumB, sqB, bn1_g, bn1_b, N);
    }

    // ---- 3. 5x GatedGraphConv steps ----------------------------------------
    for (int i = 0; i < 5; ++i) {
        gemm_h_kernel<<<gemmBlks, 256, 0, stream>>>(
            bufH, pkGgc + (size_t)i * 8 * 16 * 512, bufM, N, nStrips);
        hipMemsetAsync(bufAgg, 0, (size_t)N * HDIM * 4, stream);
        {
            size_t tot = (size_t)E * 32;
            scatter_kernel<<<(unsigned)((tot + 255) / 256), 256, 0, stream>>>(
                bufM, src, dst, bufAgg, E);
        }
        gru_kernel<<<gemmBlks, 256, 0, stream>>>(
            bufH, bufAgg, pkWih, pkWhh, gru_bih, gru_bhh, N, nStrips);
    }

    // ---- 4. BN2 + ReLU -> h2 (bufY) ----------------------------------------
    hipMemsetAsync(sumB, 0, HDIM * 4, stream);
    hipMemsetAsync(sqB,  0, HDIM * 4, stream);
    colstats_kernel<<<(N + 511) / 512, 256, 0, stream>>>(bufH, N, sumB, sqB, 512);
    {
        size_t tot = (size_t)N * HDIM;
        bnrelu_kernel<<<(unsigned)((tot + 255) / 256), 256, 0, stream>>>(
            bufH, bufY, sumB, sqB, bn2_g, bn2_b, N);
    }

    // ---- 5. per-graph mean/max pooling -------------------------------------
    hipMemsetAsync(counts, 0, NGRAPH * 4, stream);
    hipMemsetAsync(gsum,   0, (size_t)NGRAPH * HDIM * 4, stream);
    hipMemsetAsync(gmax,   0, (size_t)NGRAPH * HDIM * 4, stream);  // bits 0 == +0.0f
    count_kernel<<<(N + 255) / 256, 256, 0, stream>>>(batch, counts, N);
    {
        size_t tot = (size_t)N * HDIM;
        pool_kernel<<<(unsigned)((tot + 255) / 256), 256, 0, stream>>>(bufY, batch, gsum, gmax, N);
    }
    gcat_kernel<<<(NGRAPH * 2 * HDIM + 255) / 256, 256, 0, stream>>>(gsum, gmax, counts, gcat);

    // ---- 6. MLP head --------------------------------------------------------
    fc_kernel<<<(NGRAPH * HDIM + 255) / 256, 256, 0, stream>>>(gcat, fc1_w, fc1_b, t1, 2 * HDIM, HDIM);
    smallnorm_kernel<<<1, HDIM, 0, stream>>>(t1, bn3_g, bn3_b, HDIM);
    fc_kernel<<<(NGRAPH * 128 + 255) / 256, 256, 0, stream>>>(t1, fc2_w, fc2_b, t2, HDIM, 128);
    smallnorm_kernel<<<1, 128, 0, stream>>>(t2, bn4_g, bn4_b, 128);
    fc_kernel<<<1, 128, 0, stream>>>(t2, fc3_w, fc3_b, (float*)d_out, 128, 2);
}
